// VQLayer_51118700757613
// MI455X (gfx1250) — matrix operand order repaired and checked by reading, not compiled, not run on hardware
//
#include <hip/hip_runtime.h>

// VQ nearest-neighbor: dist = ||x||^2 + ||e||^2 - 2 x.e  -> argmin -> gather emb row.
// score = x.e - 0.5*||e||^2 (argmax == argmin dist), computed with
// v_wmma_f32_16x16x32_bf16 on an error-compensated bf16 hi/lo split (3 WMMAs ~ fp32).
// emb tiles are double-buffered in LDS via the Tensor Data Mover (tensor_load_to_lds,
// TENSORcnt), overlapping the DMA with WMMA compute.

#define BZV   16384
#define KCV   8192
#define DDV   512

typedef __attribute__((ext_vector_type(16))) __bf16 v16bf;
typedef __attribute__((ext_vector_type(8)))  float  v8f;
typedef __attribute__((ext_vector_type(4)))  unsigned int u32x4;
typedef __attribute__((ext_vector_type(8)))  int i32x8;
typedef __attribute__((ext_vector_type(4)))  int i32x4;
typedef unsigned short u16;
typedef unsigned int   u32;

// ---------- bf16 split helpers (round-to-nearest-even) ----------
__device__ __forceinline__ u16 f2bf(float f) {
  u32 u = __float_as_uint(f);
  u += 0x7FFFu + ((u >> 16) & 1u);
  return (u16)(u >> 16);
}
__device__ __forceinline__ float bf2f(u16 h) {
  return __uint_as_float((u32)h << 16);
}

struct Q2 { uint4 a, b; };
__device__ __forceinline__ v16bf mkv(uint4 a, uint4 b) {
  Q2 t{a, b};
  return __builtin_bit_cast(v16bf, t);
}

// ---------- prep: split f32 -> bf16 hi + bf16 lo residual ----------
__global__ __launch_bounds__(256) void prep_split(const float* __restrict__ src,
                                                  u16* __restrict__ hi,
                                                  u16* __restrict__ lo, int n4) {
  int i = blockIdx.x * blockDim.x + threadIdx.x;
  if (i >= n4) return;
  float4 v = reinterpret_cast<const float4*>(src)[i];
  ushort4 h, l;
  h.x = f2bf(v.x); l.x = f2bf(v.x - bf2f(h.x));
  h.y = f2bf(v.y); l.y = f2bf(v.y - bf2f(h.y));
  h.z = f2bf(v.z); l.z = f2bf(v.z - bf2f(h.z));
  h.w = f2bf(v.w); l.w = f2bf(v.w - bf2f(h.w));
  reinterpret_cast<ushort4*>(hi)[i] = h;
  reinterpret_cast<ushort4*>(lo)[i] = l;
}

// ---------- prep: ||e||^2 per code row (one wave32 per row) ----------
__global__ __launch_bounds__(256) void row_sumsq(const float* __restrict__ emb,
                                                 float* __restrict__ e2) {
  int lane = threadIdx.x & 31, wv = threadIdx.x >> 5;
  int row = blockIdx.x * 8 + wv;
  const float4* p = reinterpret_cast<const float4*>(emb + (size_t)row * DDV);
  float s = 0.f;
#pragma unroll
  for (int c = 0; c < 4; ++c) {
    float4 v = p[c * 32 + lane];
    s += v.x * v.x + v.y * v.y + v.z * v.z + v.w * v.w;
  }
#pragma unroll
  for (int off = 16; off > 0; off >>= 1) s += __shfl_xor(s, off, 32);
  if (lane == 0) e2[row] = s;
}

// ---------- main fused GEMM + argmax + gather ----------
// LDS layout (u16 units), padded row stride 520 (=512+8) -> conflict-free ds reads.
#define RSTR     520
#define XS_L_OFF (64 * RSTR)           // x-lo plane offset      (33280 u16)
#define EB_OFF   (2 * 64 * RSTR)       // emb double-buffer base (66560 u16)
#define EB_SZ    (32 * RSTR * 2)       // one buffer: 32 codes hi+lo (33280 u16)
#define EB_LO    (32 * RSTR)           // lo plane inside a buffer
#define SMEM_BYTES ((EB_OFF + 2 * EB_SZ) * 2)   // 266240 B  (< 320KB WGP LDS)

// TDM: DMA one 32-code x 512-d bf16 tile (1024B rows) into LDS with 16B row pad
// (pad_interval=7 -> every 256 DWORDs, pad_amount=3 -> 4 DWORDs) => 520-u16 stride.
__device__ __forceinline__ void tdm_load_tile(const u16* gsrc, unsigned ldsByteOff) {
  unsigned long long ga = (unsigned long long)(size_t)gsrc;
  u32x4 g0;
  g0[0] = 1u;                                   // count=1, user descriptor
  g0[1] = ldsByteOff;                           // lds_addr (bytes)
  g0[2] = (u32)(ga & 0xFFFFFFFFu);              // global_addr[31:0]
  g0[3] = (u32)((ga >> 32) & 0x01FFFFFFu)       // global_addr[56:32]
        | (2u << 30);                           // type = 2 ("image")
  i32x8 g1;
  g1[0] = (int)((1u << 16)                      // data_size = 2 bytes
              | (1u << 20)                      // pad_enable
              | (7u << 22)                      // pad_interval: 256 DWORDs
              | (3u << 25));                    // pad_amount: 4 DWORDs (16B)
  g1[1] = (int)(512u << 16);                    // tensor_dim0[15:0] in [31:16]
  g1[2] = (int)(8192u << 16);                   // tensor_dim0 hi | tensor_dim1 lo
  g1[3] = (int)(512u << 16);                    // tensor_dim1 hi | tile_dim0=512
  g1[4] = 32;                                   // tile_dim1=32 | tile_dim2=0
  g1[5] = 512;                                  // tensor_dim0_stride[31:0]
  g1[6] = 0;                                    // stride hi | dim1_stride lo
  g1[7] = 0;
  i32x4 z4 = {0, 0, 0, 0};
#if defined(__clang_major__) && (__clang_major__ >= 23)
  i32x8 z8 = {0, 0, 0, 0, 0, 0, 0, 0};
  __builtin_amdgcn_tensor_load_to_lds(g0, g1, z4, z4, z8, 0);
#else
  __builtin_amdgcn_tensor_load_to_lds(g0, g1, z4, z4, 0);
#endif
}

__launch_bounds__(128, 1)
__global__ void vq_main(const u16* __restrict__ xh, const u16* __restrict__ xl,
                        const u16* __restrict__ eh, const u16* __restrict__ el,
                        const float* __restrict__ e2, const float* __restrict__ emb,
                        float* __restrict__ out) {
  extern __shared__ u16 s[];
  const int tid  = threadIdx.x;
  const int lane = tid & 31;
  const int wv   = tid >> 5;       // 4 waves, 16 x-rows each
  const int half = lane >> 4;      // lane half selects K-chunk (ISA A/B layouts)
  const int l16  = lane & 15;
  const int rowBlk = blockIdx.x * 64;

  // kick off TDM for emb tile 0 (wave0: hi plane, wave1: lo plane)
  if (wv == 0) tdm_load_tile(eh, EB_OFF * 2);
  if (wv == 1) tdm_load_tile(el, (EB_OFF + EB_LO) * 2);

  // stage this block's 64 x rows (hi+lo bf16) into padded LDS
  {
    const size_t gbase = (size_t)rowBlk * DDV;
    for (int i = tid; i < 4096; i += 128) {
      int r = i >> 6, seg = i & 63;
      uint4 vh = *reinterpret_cast<const uint4*>(xh + gbase + r * DDV + seg * 8);
      uint4 vl = *reinterpret_cast<const uint4*>(xl + gbase + r * DDV + seg * 8);
      *reinterpret_cast<uint4*>(s + r * RSTR + seg * 8) = vh;
      *reinterpret_cast<uint4*>(s + XS_L_OFF + r * RSTR + seg * 8) = vl;
    }
  }
  __builtin_amdgcn_s_wait_tensorcnt(0);
  __syncthreads();

  float bestv = -3.4e38f;
  int   besti = 0;

  const int kb16 = half << 3;  // B operand: K-chunks at +0/+16, base 0 or 8
  const int ka8  = half << 3;  // A operand: K-chunks at +0/+16, base 0 or 8
  const u16* xsh = s + (wv * 16 + l16) * RSTR + kb16;
  const u16* xsl = xsh + XS_L_OFF;

  const int nT = KCV / 32;
  for (int t = 0; t < nT; ++t) {
    const int cb = t * 32;
    const u16* ebh = s + EB_OFF + (t & 1) * EB_SZ;
    const u16* ebl = ebh + EB_LO;

    // async prefetch of next tile via TDM while this tile computes
    if (t + 1 < nT) {
      const unsigned dstOff = (unsigned)(EB_OFF + ((t + 1) & 1) * EB_SZ) * 2u;
      if (wv == 0) tdm_load_tile(eh + (size_t)(t + 1) * 32 * DDV, dstOff);
      if (wv == 1) tdm_load_tile(el + (size_t)(t + 1) * 32 * DDV, dstOff + EB_LO * 2u);
    }

    // init accumulators with -0.5*||e||^2  (score = dot - 0.5*e2)
    v8f acc0, acc1;
    {
      const float* p0 = e2 + cb + (half << 3);
      const float* p1 = e2 + cb + 16 + (half << 3);
      float4 a0 = *reinterpret_cast<const float4*>(p0);
      float4 a1 = *reinterpret_cast<const float4*>(p0 + 4);
      float4 b0 = *reinterpret_cast<const float4*>(p1);
      float4 b1 = *reinterpret_cast<const float4*>(p1 + 4);
      acc0[0] = -0.5f * a0.x; acc0[1] = -0.5f * a0.y; acc0[2] = -0.5f * a0.z; acc0[3] = -0.5f * a0.w;
      acc0[4] = -0.5f * a1.x; acc0[5] = -0.5f * a1.y; acc0[6] = -0.5f * a1.z; acc0[7] = -0.5f * a1.w;
      acc1[0] = -0.5f * b0.x; acc1[1] = -0.5f * b0.y; acc1[2] = -0.5f * b0.z; acc1[3] = -0.5f * b0.w;
      acc1[4] = -0.5f * b1.x; acc1[5] = -0.5f * b1.y; acc1[6] = -0.5f * b1.z; acc1[7] = -0.5f * b1.w;
    }

    const u16* ea0  = ebh + l16 * RSTR + ka8;
    const u16* ea0l = ebl + l16 * RSTR + ka8;
    const u16* ea1  = ebh + (16 + l16) * RSTR + ka8;
    const u16* ea1l = ebl + (16 + l16) * RSTR + ka8;

#pragma unroll 4
    for (int d0 = 0; d0 < DDV; d0 += 32) {
      // B (x tile, 32x16): two 8-elem K-chunks per lane at +0 / +16
      uint4 q0 = *reinterpret_cast<const uint4*>(xsh + d0);
      uint4 q1 = *reinterpret_cast<const uint4*>(xsh + d0 + 16);
      v16bf Bh = mkv(q0, q1);
      q0 = *reinterpret_cast<const uint4*>(xsl + d0);
      q1 = *reinterpret_cast<const uint4*>(xsl + d0 + 16);
      v16bf Bl = mkv(q0, q1);
      // A (emb tiles, 16x32): two 8-elem K-chunks per lane at +0 / +16
      uint4 t0 = *reinterpret_cast<const uint4*>(ea0 + d0);
      uint4 t1 = *reinterpret_cast<const uint4*>(ea0 + d0 + 16);
      v16bf A0h = mkv(t0, t1);
      t0 = *reinterpret_cast<const uint4*>(ea0l + d0);
      t1 = *reinterpret_cast<const uint4*>(ea0l + d0 + 16);
      v16bf A0l = mkv(t0, t1);
      t0 = *reinterpret_cast<const uint4*>(ea1 + d0);
      t1 = *reinterpret_cast<const uint4*>(ea1 + d0 + 16);
      v16bf A1h = mkv(t0, t1);
      t0 = *reinterpret_cast<const uint4*>(ea1l + d0);
      t1 = *reinterpret_cast<const uint4*>(ea1l + d0 + 16);
      v16bf A1l = mkv(t0, t1);
      // bf16x3 compensated dot: hi*hi + hi*lo + lo*hi, two independent chains
      acc0 = __builtin_amdgcn_wmma_f32_16x16x32_bf16(false, A0h, false, Bh, (short)0, acc0, false, false);
      acc1 = __builtin_amdgcn_wmma_f32_16x16x32_bf16(false, A1h, false, Bh, (short)0, acc1, false, false);
      acc0 = __builtin_amdgcn_wmma_f32_16x16x32_bf16(false, A0h, false, Bl, (short)0, acc0, false, false);
      acc1 = __builtin_amdgcn_wmma_f32_16x16x32_bf16(false, A1h, false, Bl, (short)0, acc1, false, false);
      acc0 = __builtin_amdgcn_wmma_f32_16x16x32_bf16(false, A0l, false, Bh, (short)0, acc0, false, false);
      acc1 = __builtin_amdgcn_wmma_f32_16x16x32_bf16(false, A1l, false, Bh, (short)0, acc1, false, false);
    }

    // C layout: VGPR r -> code M = r + 8*half; lane -> x row. Running argmax.
#pragma unroll
    for (int r = 0; r < 8; ++r) {
      float v0 = acc0[r];
      int   c0 = cb + (half << 3) + r;
      if (v0 > bestv) { bestv = v0; besti = c0; }
    }
#pragma unroll
    for (int r = 0; r < 8; ++r) {
      float v1 = acc1[r];
      int   c1 = cb + 16 + (half << 3) + r;
      if (v1 > bestv) { bestv = v1; besti = c1; }
    }

    // next tile's DMA must be complete before anyone reads it
    __builtin_amdgcn_s_wait_tensorcnt(0);
    __syncthreads();
  }

  // combine lane n with lane n+16 (they hold the same x row, different codes)
  {
    float ov = __shfl_xor(bestv, 16, 32);
    int   oi = __shfl_xor(besti, 16, 32);
    if (ov > bestv || (ov == bestv && oi < besti)) { bestv = ov; besti = oi; }
  }

  // gather emb[best] -> out, whole wave per row, float4 vectorized
  const int row0 = rowBlk + wv * 16;
  for (int i = 0; i < 16; ++i) {
    int idx = __shfl(besti, i, 32);
    const float4* srcp = reinterpret_cast<const float4*>(emb + (size_t)idx * DDV);
    float4* dstp = reinterpret_cast<float4*>(out + (size_t)(row0 + i) * DDV);
#pragma unroll
    for (int j = 0; j < 4; ++j) dstp[lane + 32 * j] = srcp[lane + 32 * j];
  }
}

extern "C" void kernel_launch(void* const* d_in, const int* in_sizes, int n_in,
                              void* d_out, int out_size, void* d_ws, size_t ws_size,
                              hipStream_t stream) {
  (void)in_sizes; (void)n_in; (void)out_size; (void)ws_size;
  const float* x   = (const float*)d_in[0];
  const float* emb = (const float*)d_in[1];
  float* out = (float*)d_out;

  // workspace carve (~50.4 MB): x hi/lo, emb hi/lo (bf16), e2 (f32)
  u16* xh = (u16*)d_ws;
  u16* xl = xh + (size_t)BZV * DDV;
  u16* eh = xl + (size_t)BZV * DDV;
  u16* el = eh + (size_t)KCV * DDV;
  float* e2 = (float*)(el + (size_t)KCV * DDV);

  hipFuncSetAttribute((const void*)vq_main,
                      hipFuncAttributeMaxDynamicSharedMemorySize, SMEM_BYTES);

  int n4x = BZV * DDV / 4;
  prep_split<<<n4x / 256, 256, 0, stream>>>(x, xh, xl, n4x);
  int n4e = KCV * DDV / 4;
  prep_split<<<n4e / 256, 256, 0, stream>>>(emb, eh, el, n4e);
  row_sumsq<<<KCV / 8, 256, 0, stream>>>(emb, e2);

  vq_main<<<BZV / 64, 128, SMEM_BYTES, stream>>>(xh, xl, eh, el, e2, emb, out);
}
